// Cross_attention_78778290144033
// MI455X (gfx1250) — matrix operand order repaired and checked by reading, not compiled, not run on hardware
//
#include <hip/hip_runtime.h>
#include <hip/hip_bf16.h>
#include <math.h>

#define DI __device__ __forceinline__

constexpr int   kB   = 2;
constexpr int   kC   = 128;
constexpr int   kL   = 64 * 64;       // H*W = 4096
constexpr int   kNH  = 4;
constexpr int   kD   = kC / kNH;      // 32
constexpr int   kG   = 16;
constexpr int   kCpG = kC / kG;       // 8
constexpr float kEPS   = 1e-5f;
constexpr float kScale = 0.08838834764831845f; // 1/sqrt(C)

typedef __attribute__((ext_vector_type(16))) __bf16 v16bf;
typedef __attribute__((ext_vector_type(8)))  float  v8f;

union U16x16 { uint4 q[2]; v16bf v; __bf16 b[16]; };
union U16x8  { uint4 q;             __bf16 b[8];  };

// D = A(16x32 bf16) * B(32x16 bf16) + C(16x16 f32)
DI v8f wmma_bf16(v16bf a, v16bf b, v8f c) {
  return __builtin_amdgcn_wmma_f32_16x16x32_bf16(
      /*neg_a=*/false, a, /*neg_b=*/false, b,
      /*c_mod=*/(short)0, c, /*reuse_a=*/false, /*reuse_b=*/false);
}

// Two contiguous 16B loads -> one 16-element bf16 fragment
DI v16bf load16(const __bf16* p0, const __bf16* p1) {
  U16x16 u;
  u.q[0] = *(const uint4*)p0;
  u.q[1] = *(const uint4*)p1;
  return u.v;
}

// Convert 16 f32 (at rw[0..15]) to an A fragment
DI v16bf cvt_a_f32(const float* rw) {
  float4 a0 = *(const float4*)(rw);
  float4 a1 = *(const float4*)(rw + 4);
  float4 a2 = *(const float4*)(rw + 8);
  float4 a3 = *(const float4*)(rw + 12);
  U16x16 u;
  u.b[0]  = (__bf16)a0.x; u.b[1]  = (__bf16)a0.y; u.b[2]  = (__bf16)a0.z; u.b[3]  = (__bf16)a0.w;
  u.b[4]  = (__bf16)a1.x; u.b[5]  = (__bf16)a1.y; u.b[6]  = (__bf16)a1.z; u.b[7]  = (__bf16)a1.w;
  u.b[8]  = (__bf16)a2.x; u.b[9]  = (__bf16)a2.y; u.b[10] = (__bf16)a2.z; u.b[11] = (__bf16)a2.w;
  u.b[12] = (__bf16)a3.x; u.b[13] = (__bf16)a3.y; u.b[14] = (__bf16)a3.z; u.b[15] = (__bf16)a3.w;
  return u.v;
}

// exchange with lane^16 via ds_swizzle SWAPX16 (xor=0x10, and=0x1f)
DI float xor16(float x) {
  return __int_as_float(__builtin_amdgcn_ds_swizzle(__float_as_int(x), 0x401F));
}

// ---------------------------------------------------------------------------
// Kernel 1: GroupNorm for both streams.  grid (kB*kG, 2), block 256
// ---------------------------------------------------------------------------
__global__ void gn_kernel(const float* __restrict__ xA, const float* __restrict__ xB,
                          const float* __restrict__ wA, const float* __restrict__ bA,
                          const float* __restrict__ wB, const float* __restrict__ bB,
                          float* __restrict__ xnA, float* __restrict__ xnB) {
  const int tid = threadIdx.x;
  const int b = blockIdx.x / kG, g = blockIdx.x % kG;
  const bool sB = (blockIdx.y == 1);
  const float* x  = sB ? xB : xA;
  const float* w  = sB ? wB : wA;
  const float* bb = sB ? bB : bA;
  float* xn       = sB ? xnB : xnA;

  const int ch0 = g * kCpG;
  const size_t base = ((size_t)b * kC + ch0) * kL;
  const int n = kCpG * kL; // 32768

  float s = 0.f, ss = 0.f;
  for (int i = tid; i < n; i += 256) {
    float v = x[base + i];
    s += v; ss += v * v;
  }
  __shared__ float rs[256], rq[256];
  rs[tid] = s; rq[tid] = ss;
  __syncthreads();
  for (int st = 128; st > 0; st >>= 1) {
    if (tid < st) { rs[tid] += rs[tid + st]; rq[tid] += rq[tid + st]; }
    __syncthreads();
  }
  const float inv_n = 1.f / (float)n;
  const float mean = rs[0] * inv_n;
  const float var  = rq[0] * inv_n - mean * mean;
  const float rstd = rsqrtf(var + kEPS);

  for (int i = tid; i < n; i += 256) {
    int c = ch0 + (i >> 12);             // i / 4096
    float v = x[base + i];
    xn[base + i] = (v - mean) * rstd * w[c] + bb[c];
  }
}

// ---------------------------------------------------------------------------
// Kernel 2: QKV projection  [384x128] @ xn[128 x L]  -> qT/kT/v (bf16)
// qT,kT: [b][h][pos][32]   v: [b][h][32][pos]       grid (L/128, kB), block 256
// ---------------------------------------------------------------------------
__global__ void qkv_kernel(const float* __restrict__ xn, const float* __restrict__ W,
                           __bf16* __restrict__ qT, __bf16* __restrict__ kT,
                           __bf16* __restrict__ vv) {
  const int tid = threadIdx.x;
  const int lane = tid & 31, wave = tid >> 5;
  const int half = lane >> 4, n = lane & 15;
  const int b = blockIdx.y;
  const int col = blockIdx.x * 128 + wave * 16 + n;

  // B fragments: xn[K, col] for K-steps of 32  (element e = K 16*half+e)
  v16bf bq[4];
#pragma unroll
  for (int ks = 0; ks < 4; ++ks) {
    U16x16 u;
    const float* src = xn + ((size_t)b * kC + ks * 32 + 16 * half) * kL + col;
#pragma unroll
    for (int e = 0; e < 16; ++e) u.b[e] = (__bf16)src[(size_t)e * kL];
    bq[ks] = u.v;
  }

#pragma unroll
  for (int mt = 0; mt < 24; ++mt) {
    v8f acc = 0.f;
#pragma unroll
    for (int ks = 0; ks < 4; ++ks) {
      const float* rw = W + (size_t)(mt * 16 + n) * kC + ks * 32 + 16 * half;
      acc = wmma_bf16(cvt_a_f32(rw), bq[ks], acc);
    }
    // row group: head h, part (0=q,1=k,2=v), channel base within head
    const int h = mt / 6, sub = mt % 6, part = sub >> 1, cb = (sub & 1) * 16;
    if (part == 2) {
      // v: [b][h][chan][L]
#pragma unroll
      for (int r = 0; r < 8; ++r) {
        int chan = cb + 8 * half + r;
        vv[(((size_t)b * kNH + h) * kD + chan) * kL + col] = (__bf16)acc[r];
      }
    } else {
      __bf16* dst = (part == 0 ? qT : kT) +
                    (((size_t)b * kNH + h) * kL + col) * kD + cb + 8 * half;
      const float sc = (part == 0) ? kScale : 1.0f;
      U16x8 st;
#pragma unroll
      for (int r = 0; r < 8; ++r) st.b[r] = (__bf16)(acc[r] * sc);
      *(uint4*)dst = st.q;
    }
  }
}

// ---------------------------------------------------------------------------
// Kernel 3: flash cross-attention, transposed formulation Sᵀ = Kᵀ·Q.
// outT: [b][pos][C] bf16.   grid (L/128, kNH, kB), block 256 (8 waves)
// ---------------------------------------------------------------------------
__global__ void attn_kernel(const __bf16* __restrict__ qT, const __bf16* __restrict__ kT,
                            const __bf16* __restrict__ vv, __bf16* __restrict__ outT) {
  const int tid = threadIdx.x, lane = tid & 31, wave = tid >> 5;
  const int half = lane >> 4, n = lane & 15;
  const int h = blockIdx.y, b = blockIdx.z;
  const int qbase = (blockIdx.x * 8 + wave) * 16;
  const size_t bh = (size_t)b * kNH + h;

  // fixed B operand: 16 queries x 32 channels (element e = chan 8*half+e, e>=8: 16+8*half+e-8)
  const __bf16* qrow = qT + (bh * kL + qbase + n) * kD + 8 * half;
  const v16bf bq = load16(qrow, qrow + 16);

  const __bf16* kb_p = kT + bh * kL * kD;
  const __bf16* vb_p = vv + bh * kD * kL;

  v8f o_lo = 0.f, o_hi = 0.f;
  float m_run = -INFINITY, l_run = 0.f;
  const v8f zero = 0.f;

  for (int kc = 0; kc < kL / 32; ++kc) {
    const int kb = kc * 32;
    // scores: A = kT rows (keys), K dim = 32 channels
    const __bf16* kr0 = kb_p + (size_t)(kb + n) * kD + 8 * half;
    const __bf16* kr1 = kb_p + (size_t)(kb + 16 + n) * kD + 8 * half;
    v8f s_lo = wmma_bf16(load16(kr0, kr0 + 16), bq, zero); // keys kb..kb+15
    v8f s_hi = wmma_bf16(load16(kr1, kr1 + 16), bq, zero); // keys kb+16..kb+31

    // per-lane: 16 score values of query column n; partner lane^16 has other 16
    float mx = s_lo[0];
#pragma unroll
    for (int r = 1; r < 8; ++r) mx = fmaxf(mx, s_lo[r]);
#pragma unroll
    for (int r = 0; r < 8; ++r) mx = fmaxf(mx, s_hi[r]);
    mx = fmaxf(mx, xor16(mx));
    const float m_new = fmaxf(m_run, mx);
    const float alpha = __expf(m_run - m_new);

    float pl[8], ph[8], ssum = 0.f;
#pragma unroll
    for (int r = 0; r < 8; ++r) { pl[r] = __expf(s_lo[r] - m_new); ssum += pl[r]; }
#pragma unroll
    for (int r = 0; r < 8; ++r) { ph[r] = __expf(s_hi[r] - m_new); ssum += ph[r]; }
    ssum += xor16(ssum);
    l_run = l_run * alpha + ssum;
#pragma unroll
    for (int r = 0; r < 8; ++r) { o_lo[r] *= alpha; o_hi[r] *= alpha; }

    // build Pᵀ B-fragment: element e = P[key 16*half+e, query n]
    U16x16 pb;
#pragma unroll
    for (int e = 0; e < 8; ++e) {
      float plp = xor16(pl[e]);   // partner's keys 8*(1-half)+e
      float php = xor16(ph[e]);   // partner's keys 16+8*(1-half)+e
      pb.b[e]     = (__bf16)(half ? php   : pl[e]);
      pb.b[8 + e] = (__bf16)(half ? ph[e] : plp);
    }

    // PV: A = v rows (d channels), K dim = 32 keys of this chunk
    const __bf16* vr0 = vb_p + (size_t)n * kL + kb + 16 * half;
    const __bf16* vr1 = vb_p + (size_t)(16 + n) * kL + kb + 16 * half;
    o_lo = wmma_bf16(load16(vr0, vr0 + 8), pb.v, o_lo);
    o_hi = wmma_bf16(load16(vr1, vr1 + 8), pb.v, o_hi);

    m_run = m_new;
  }

  const float inv = 1.0f / l_run;
  __bf16* drow = outT + ((size_t)b * kL + qbase + n) * kC + h * kD;
  U16x8 st;
#pragma unroll
  for (int r = 0; r < 8; ++r) st.b[r] = (__bf16)(o_lo[r] * inv);
  *(uint4*)(drow + 8 * half) = st.q;
#pragma unroll
  for (int r = 0; r < 8; ++r) st.b[r] = (__bf16)(o_hi[r] * inv);
  *(uint4*)(drow + 16 + 8 * half) = st.q;
}

// ---------------------------------------------------------------------------
// Kernel 4: out projection + bias + residual (f32 out). grid (L/128, kB)
// ---------------------------------------------------------------------------
__global__ void proj_kernel(const __bf16* __restrict__ aT, const float* __restrict__ W,
                            const float* __restrict__ bias, const float* __restrict__ xn,
                            float* __restrict__ out) {
  const int tid = threadIdx.x, lane = tid & 31, wave = tid >> 5;
  const int half = lane >> 4, n = lane & 15;
  const int b = blockIdx.y;
  const int col = blockIdx.x * 128 + wave * 16 + n;

  v16bf bx[4];
  const __bf16* arow = aT + ((size_t)b * kL + col) * kC;
#pragma unroll
  for (int ks = 0; ks < 4; ++ks) {
    const __bf16* p = arow + ks * 32 + 16 * half;
    bx[ks] = load16(p, p + 8);
  }

#pragma unroll
  for (int mt = 0; mt < 8; ++mt) {
    v8f acc = 0.f;
#pragma unroll
    for (int ks = 0; ks < 4; ++ks) {
      const float* rw = W + (size_t)(mt * 16 + n) * kC + ks * 32 + 16 * half;
      acc = wmma_bf16(cvt_a_f32(rw), bx[ks], acc);
    }
#pragma unroll
    for (int r = 0; r < 8; ++r) {
      int row = mt * 16 + 8 * half + r;
      size_t idx = ((size_t)b * kC + row) * kL + col;
      out[idx] = acc[r] + bias[row] + xn[idx];
    }
  }
}

// ---------------------------------------------------------------------------
extern "C" void kernel_launch(void* const* d_in, const int* in_sizes, int n_in,
                              void* d_out, int out_size, void* d_ws, size_t ws_size,
                              hipStream_t stream) {
  (void)in_sizes; (void)n_in; (void)out_size; (void)ws_size;
  const float* xA    = (const float*)d_in[0];
  const float* xB    = (const float*)d_in[1];
  const float* gwA   = (const float*)d_in[2];
  const float* gbA   = (const float*)d_in[3];
  const float* gwB   = (const float*)d_in[4];
  const float* gbB   = (const float*)d_in[5];
  const float* wqkvA = (const float*)d_in[6];
  const float* wqkvB = (const float*)d_in[7];
  const float* woutA = (const float*)d_in[8];
  const float* boutA = (const float*)d_in[9];
  const float* woutB = (const float*)d_in[10];
  const float* boutB = (const float*)d_in[11];
  float* out = (float*)d_out;

  const size_t szXn = (size_t)kB * kC * kL;          // f32 elements (1M)
  float* xnA = (float*)d_ws;
  float* xnB = xnA + szXn;
  __bf16* p  = (__bf16*)(xnB + szXn);
  const size_t szQ = (size_t)kB * kNH * kL * kD;     // bf16 elements (1M)
  __bf16* qTA = p + 0 * szQ; __bf16* kTA = p + 1 * szQ; __bf16* vA = p + 2 * szQ;
  __bf16* qTB = p + 3 * szQ; __bf16* kTB = p + 4 * szQ; __bf16* vB = p + 5 * szQ;
  __bf16* aoA = p + 6 * szQ; __bf16* aoB = p + 7 * szQ;

  gn_kernel<<<dim3(kB * kG, 2), 256, 0, stream>>>(xA, xB, gwA, gbA, gwB, gbB, xnA, xnB);

  qkv_kernel<<<dim3(kL / 128, kB), 256, 0, stream>>>(xnA, wqkvA, qTA, kTA, vA);
  qkv_kernel<<<dim3(kL / 128, kB), 256, 0, stream>>>(xnB, wqkvB, qTB, kTB, vB);

  // out_A: queries from B attend to K/V of A (and vice versa)
  attn_kernel<<<dim3(kL / 128, kNH, kB), 256, 0, stream>>>(qTB, kTA, vA, aoA);
  attn_kernel<<<dim3(kL / 128, kNH, kB), 256, 0, stream>>>(qTA, kTB, vB, aoB);

  proj_kernel<<<dim3(kL / 128, kB), 256, 0, stream>>>(aoA, woutA, boutA, xnA, out);
  proj_kernel<<<dim3(kL / 128, kB), 256, 0, stream>>>(aoB, woutB, boutB, xnB, out + szXn);
}
